// Swin_Block_44375602102366
// MI455X (gfx1250) — matrix operand order repaired and checked by reading, not compiled, not run on hardware
//
#include <hip/hip_runtime.h>
#include <math.h>

// ---------------------------------------------------------------------------
// Swin block for MI455X (gfx1250): WMMA f16->f32 GEMMs with async-LDS double
// buffering + fused LN/attn/dwconv
// ---------------------------------------------------------------------------

typedef __attribute__((ext_vector_type(16))) _Float16 v16h;
typedef __attribute__((ext_vector_type(8)))  float    v8f;

union H16 { v16h v; uint4 u[2]; };

#define DIMC   384
#define HEADS  12
#define HD     32
#define WSZ    7
#define LWIN   49          // WSZ*WSZ
#define HH     56
#define NSP    3136        // 56*56
#define BB     16
#define NWIN   64          // (56/7)^2
#define MROWS  50176       // BB*NSP == BB*NWIN*LWIN
#define HID    1536        // 4*DIMC

#define BM 128
#define BN 128
#define BK 32
#define ATILE (BM*BK)          // halves
#define BTILE (BN*BK)          // halves
#define BUFH  (ATILE + BTILE)  // halves per double-buffer stage

// gfx1250 async copy: global -> LDS, 16B per lane, tracked by ASYNCcnt
__device__ __forceinline__ void async_b128(unsigned lds_byte, const _Float16* g) {
    asm volatile("global_load_async_to_lds_b128 %0, %1, off"
                 :: "v"(lds_byte), "v"((unsigned long long)(size_t)g)
                 : "memory");
}

// ---------------------------------------------------------------------------
// f32 -> f16 weight conversion with transpose: src (K,N) row-major ->
// dst (N,K) row-major, so GEMM B tiles are contiguous K-chunks per N-row.
// ---------------------------------------------------------------------------
__global__ void cvt_transpose(const float* __restrict__ s, _Float16* __restrict__ d,
                              int K, int N) {
    int idx = blockIdx.x * blockDim.x + threadIdx.x;
    if (idx < K * N) {
        int k = idx / N, n = idx % N;
        d[(size_t)n * K + k] = (_Float16)s[idx];
    }
}

// ---------------------------------------------------------------------------
// LayerNorm (+ optional shift-roll + window-partition gather), f16 output
// ---------------------------------------------------------------------------
template<int MODE>
__global__ __launch_bounds__(384)
void ln_kernel(const float* __restrict__ x, const float* __restrict__ g,
               const float* __restrict__ bta, _Float16* __restrict__ out) {
    __shared__ float red[512];
    __shared__ float red2[512];
    const int r = blockIdx.x;
    const int c = threadIdx.x;

    size_t src;
    if (MODE == 0) {
        src = (size_t)r * DIMC + c;
    } else {
        int win = r / LWIN, l = r % LWIN;
        int b  = win >> 6, w6 = win & 63;
        int wh = w6 >> 3,  ww = w6 & 7;
        int i  = l / WSZ,  j  = l % WSZ;
        int sh = (wh * WSZ + i + 3) % HH;    // roll(-3)
        int sw = (ww * WSZ + j + 3) % HH;
        src = ((size_t)b * NSP + sh * HH + sw) * DIMC + c;
    }
    float v = x[src];
    if (c < 128) { red[384 + c] = 0.f; red2[384 + c] = 0.f; }
    red[c] = v; red2[c] = v * v;
    __syncthreads();
    for (int s = 256; s > 0; s >>= 1) {
        if (c < s) { red[c] += red[c + s]; red2[c] += red2[c + s]; }
        __syncthreads();
    }
    float mean = red[0] * (1.f / DIMC);
    float var  = red2[0] * (1.f / DIMC) - mean * mean;
    float rstd = rsqrtf(var + 1e-5f);
    out[(size_t)r * DIMC + c] = (_Float16)((v - mean) * rstd * g[c] + bta[c]);
}

// ---------------------------------------------------------------------------
// f16 x f16(pre-transposed) -> f32 WMMA GEMM. 128x128x32 block tile, 8 waves,
// wave tile 64x32. Tiles staged with global_load_async_to_lds_b128 into a
// double-buffered LDS (A as [m][k], B as [n][k]); next tile prefetches under
// the current tile's WMMAs (s_wait_asynccnt 4 retires only the older stage).
// Epilogues: 0 = f16 store (qkv)
//            1 = +bias +window-reverse scatter +residual(x), f32 (proj)
//            2 = +bias, f16 store (fc1)
//            3 = +bias +residual, f32 store (fc2 -> d_out)
// ---------------------------------------------------------------------------
template<int EPI>
__global__ __launch_bounds__(256)
void gemm_f16(const _Float16* __restrict__ A, const _Float16* __restrict__ Bt,
              void* __restrict__ D, const float* __restrict__ bias,
              const float* __restrict__ resid, int M, int N, int K) {
    __shared__ alignas(16) _Float16 smem[2 * BUFH];

    const int tid  = threadIdx.x;
    const int lane = tid & 31;
    const int wid  = tid >> 5;
    const int wm   = wid & 1;         // row offset wm*64
    const int wn   = wid >> 1;        // col offset wn*32
    const int l16  = lane & 15;
    const int lh   = lane >> 4;
    const int bm   = blockIdx.y * BM;
    const int bn   = blockIdx.x * BN;

    // per-thread async staging slots: thread -> (row = tid/4, 8-half chunk)
    const int srow = tid >> 2;                 // 0..63
    const int sko  = (tid & 3) << 3;           // half offset 0/8/16/24
    const _Float16* gA = A  + (size_t)(bm + srow) * K + sko;
    const _Float16* gB = Bt + (size_t)(bn + srow) * K + sko;
    const unsigned smem_base = (unsigned)(size_t)smem;           // LDS byte offset
    const unsigned lA = (unsigned)(srow * BK + sko) * 2;
    const unsigned lB = (unsigned)(ATILE + srow * BK + sko) * 2;

    v8f zero = {0.f,0.f,0.f,0.f,0.f,0.f,0.f,0.f};
    v8f acc[4][2];
    #pragma unroll
    for (int mt = 0; mt < 4; ++mt)
        #pragma unroll
        for (int nt = 0; nt < 2; ++nt) acc[mt][nt] = zero;

    auto issue = [&](int buf, int kk) {
        const size_t ka = (size_t)kk * BK;
        const unsigned bb = smem_base + (unsigned)(buf * BUFH * 2);
        async_b128(bb + lA,                gA + ka);
        async_b128(bb + lA + 64 * BK * 2,  gA + ka + (size_t)64 * K);
        async_b128(bb + lB,                gB + ka);
        async_b128(bb + lB + 64 * BK * 2,  gB + ka + (size_t)64 * K);
    };

    const int nk = K / BK;
    issue(0, 0);

    for (int kk = 0; kk < nk; ++kk) {
        const int cur = kk & 1;
        if (kk + 1 < nk) {
            issue(cur ^ 1, kk + 1);
            asm volatile("s_wait_asynccnt 4" ::: "memory");   // older stage done
        } else {
            asm volatile("s_wait_asynccnt 0" ::: "memory");
        }
        __syncthreads();

        const _Float16* As = smem + cur * BUFH;
        const _Float16* Bs = As + ATILE;

        H16 a[4], b[2];
        #pragma unroll
        for (int mt = 0; mt < 4; ++mt) {
            int row = wm * 64 + mt * 16 + l16;
            int kb  = lh * 8;                 // A layout: K = h + (h>=8?8:0) + lh*8
            a[mt].u[0] = *(const uint4*)(As + row * BK + kb);
            a[mt].u[1] = *(const uint4*)(As + row * BK + 16 + kb);
        }
        #pragma unroll
        for (int nt = 0; nt < 2; ++nt) {
            int nn = wn * 32 + nt * 16 + l16;
            int kb = lh * 8;                  // B layout: K = h + (h>=8?8:0) + lh*8
            b[nt].u[0] = *(const uint4*)(Bs + nn * BK + kb);
            b[nt].u[1] = *(const uint4*)(Bs + nn * BK + kb + 16);
        }
        #pragma unroll
        for (int mt = 0; mt < 4; ++mt)
            #pragma unroll
            for (int nt = 0; nt < 2; ++nt)
                acc[mt][nt] = __builtin_amdgcn_wmma_f32_16x16x32_f16(
                    false, a[mt].v, false, b[nt].v, (short)0, acc[mt][nt], false, false);
        __syncthreads();   // all reads of `cur` done before it is overwritten
    }

    // Epilogue: C layout -> M = v + 8*lh, N = l16
    #pragma unroll
    for (int mt = 0; mt < 4; ++mt) {
        #pragma unroll
        for (int nt = 0; nt < 2; ++nt) {
            #pragma unroll
            for (int v = 0; v < 8; ++v) {
                int r  = bm + wm * 64 + mt * 16 + v + 8 * lh;
                int cc = bn + wn * 32 + nt * 16 + l16;
                float val = acc[mt][nt][v];
                if (EPI == 0) {
                    ((_Float16*)D)[(size_t)r * N + cc] = (_Float16)val;
                } else if (EPI == 1) {
                    int win = r / LWIN, l = r % LWIN;
                    int bidx = win >> 6, w6 = win & 63;
                    int wh = w6 >> 3, ww = w6 & 7;
                    int i = l / WSZ, j = l % WSZ;
                    int sp = (wh * WSZ + i) * HH + (ww * WSZ + j);
                    size_t o = ((size_t)bidx * NSP + sp) * DIMC + cc;
                    ((float*)D)[o] = val + bias[cc] + resid[o];
                } else if (EPI == 2) {
                    ((_Float16*)D)[(size_t)r * N + cc] = (_Float16)(val + bias[cc]);
                } else {
                    size_t o = (size_t)r * DIMC + cc;
                    ((float*)D)[o] = val + bias[cc] + resid[o];
                }
            }
        }
    }
}

// ---------------------------------------------------------------------------
// Windowed attention: one block (4 waves) per (window, head).
// L=49 padded to 64. logits via WMMA, analytic shift-mask, softmax via
// half-wave shuffles, attn@v via WMMA over K=64 (2 steps).
// ---------------------------------------------------------------------------
__device__ __forceinline__ int reg56(int h) { return (h < 49) ? 0 : ((h < 53) ? 1 : 2); }

__global__ __launch_bounds__(128)
void attn_kernel(const _Float16* __restrict__ qkv, _Float16* __restrict__ outb) {
    __shared__ alignas(16) _Float16 Qs[64 * 32];
    __shared__ alignas(16) _Float16 Ks[64 * 32];
    __shared__ alignas(16) _Float16 VTs[32 * 64];   // [d][j]
    __shared__ alignas(16) _Float16 Ps[64 * 64];    // attn probs

    const int tid  = threadIdx.x;
    const int lane = tid & 31;
    const int wid  = tid >> 5;          // 0..3
    const int l16  = lane & 15;
    const int lh   = lane >> 4;

    const int head = blockIdx.x % HEADS;
    const int win  = blockIdx.x / HEADS;
    const int w64  = win & 63;
    const int wh   = w64 >> 3, ww = w64 & 7;

    const size_t base = (size_t)win * LWIN * (3 * DIMC) + head * HD;

    // stage q, k, v (v transposed) into LDS; pad rows 49..63 with zero
    for (int c = tid; c < 256; c += 128) {
        int l   = c >> 2;
        int off = (c & 3) << 3;
        uint4 dq = {0,0,0,0}, dk = {0,0,0,0}, dv = {0,0,0,0};
        if (l < LWIN) {
            const _Float16* p = qkv + base + (size_t)l * (3 * DIMC);
            dq = *(const uint4*)(p + off);
            dk = *(const uint4*)(p + DIMC + off);
            dv = *(const uint4*)(p + 2 * DIMC + off);
        }
        *(uint4*)(Qs + l * 32 + off) = dq;
        *(uint4*)(Ks + l * 32 + off) = dk;
        const _Float16* hv = (const _Float16*)&dv;
        #pragma unroll
        for (int e = 0; e < 8; ++e) VTs[(off + e) * 64 + l] = hv[e];
    }
    __syncthreads();

    v8f zero = {0.f,0.f,0.f,0.f,0.f,0.f,0.f,0.f};

    // logits strip: rows wid*16..wid*16+15, all 64 cols (4 tiles), K=32
    H16 aq;
    {
        int row = wid * 16 + l16;
        int kb  = lh * 8;
        aq.u[0] = *(const uint4*)(Qs + row * 32 + kb);
        aq.u[1] = *(const uint4*)(Qs + row * 32 + 16 + kb);
    }
    v8f lg[4];
    #pragma unroll
    for (int jt = 0; jt < 4; ++jt) {
        H16 bk;
        int nn = jt * 16 + l16;
        int kb = lh * 8;
        bk.u[0] = *(const uint4*)(Ks + nn * 32 + kb);
        bk.u[1] = *(const uint4*)(Ks + nn * 32 + kb + 16);
        lg[jt] = __builtin_amdgcn_wmma_f32_16x16x32_f16(
            false, aq.v, false, bk.v, (short)0, zero, false, false);
    }

    // scale + shift-window mask + softmax (row lives in one 16-lane half)
    const float scale = 0.1767766952966369f;  // 32^-0.5
    #pragma unroll
    for (int v = 0; v < 8; ++v) {
        int i = wid * 16 + v + 8 * lh;
        bool rowok = (i < LWIN);
        int ri = 0;
        if (rowok) ri = reg56(wh * WSZ + i / WSZ) * 3 + reg56(ww * WSZ + i % WSZ);
        float vals[4];
        #pragma unroll
        for (int jt = 0; jt < 4; ++jt) {
            int j = jt * 16 + l16;
            float xv = lg[jt][v] * scale;
            if (rowok && j < LWIN) {
                int rj = reg56(wh * WSZ + j / WSZ) * 3 + reg56(ww * WSZ + j % WSZ);
                if (ri != rj) xv -= 100.f;
            } else {
                xv = -1e30f;
            }
            vals[jt] = xv;
        }
        float m = fmaxf(fmaxf(vals[0], vals[1]), fmaxf(vals[2], vals[3]));
        #pragma unroll
        for (int s = 1; s < 16; s <<= 1) m = fmaxf(m, __shfl_xor(m, s, 32));
        float ssum = 0.f;
        #pragma unroll
        for (int jt = 0; jt < 4; ++jt) { vals[jt] = __expf(vals[jt] - m); ssum += vals[jt]; }
        #pragma unroll
        for (int s = 1; s < 16; s <<= 1) ssum += __shfl_xor(ssum, s, 32);
        float inv = 1.f / ssum;
        #pragma unroll
        for (int jt = 0; jt < 4; ++jt)
            Ps[i * 64 + jt * 16 + l16] = (_Float16)(vals[jt] * inv);
    }
    __syncthreads();

    // out strip 16x32 = attn(16x64) @ v(64x32), two K=32 steps
    v8f o[2] = {zero, zero};
    #pragma unroll
    for (int ks = 0; ks < 2; ++ks) {
        H16 ap;
        int row = wid * 16 + l16;
        int kb  = ks * 32 + lh * 8;
        ap.u[0] = *(const uint4*)(Ps + row * 64 + kb);
        ap.u[1] = *(const uint4*)(Ps + row * 64 + kb + 16);
        #pragma unroll
        for (int nt = 0; nt < 2; ++nt) {
            H16 bv;
            int nn = nt * 16 + l16;
            int kb2 = ks * 32 + lh * 8;
            bv.u[0] = *(const uint4*)(VTs + nn * 64 + kb2);
            bv.u[1] = *(const uint4*)(VTs + nn * 64 + kb2 + 16);
            o[nt] = __builtin_amdgcn_wmma_f32_16x16x32_f16(
                false, ap.v, false, bv.v, (short)0, o[nt], false, false);
        }
    }

    // store to (win*49+l, head*32+d) layout, f16
    #pragma unroll
    for (int nt = 0; nt < 2; ++nt) {
        #pragma unroll
        for (int v = 0; v < 8; ++v) {
            int i = wid * 16 + v + 8 * lh;
            if (i < LWIN)
                outb[((size_t)win * LWIN + i) * DIMC + head * HD + nt * 16 + l16] =
                    (_Float16)o[nt][v];
        }
    }
}

// ---------------------------------------------------------------------------
// Depthwise 3x3 conv + bias + exact GELU (erf), f16 in/out, f32 math
// ---------------------------------------------------------------------------
__global__ __launch_bounds__(256)
void dwconv_gelu(const _Float16* __restrict__ xin, const float* __restrict__ k9,
                 const float* __restrict__ bias, _Float16* __restrict__ out) {
    size_t idx = (size_t)blockIdx.x * 256 + threadIdx.x;
    if (idx >= (size_t)BB * NSP * HID) return;
    int c = (int)(idx % HID);
    size_t t = idx / HID;
    int n = (int)(t % NSP);
    int b = (int)(t / NSP);
    int h = n / HH, w = n % HH;
    float acc = bias[c];
    #pragma unroll
    for (int dy = 0; dy < 3; ++dy) {
        int hh = h + dy - 1;
        if (hh < 0 || hh >= HH) continue;
        #pragma unroll
        for (int dx = 0; dx < 3; ++dx) {
            int wq = w + dx - 1;
            if (wq < 0 || wq >= HH) continue;
            acc += (float)xin[((size_t)b * NSP + hh * HH + wq) * HID + c] *
                   k9[c * 9 + dy * 3 + dx];
        }
    }
    float gv = 0.5f * acc * (1.f + erff(acc * 0.70710678118654752f));
    out[idx] = (_Float16)gv;
}

// ---------------------------------------------------------------------------
// Host-side launcher
// ---------------------------------------------------------------------------
extern "C" void kernel_launch(void* const* d_in, const int* in_sizes, int n_in,
                              void* d_out, int out_size, void* d_ws, size_t ws_size,
                              hipStream_t stream) {
    (void)in_sizes; (void)n_in; (void)out_size; (void)ws_size;
    const float* x      = (const float*)d_in[0];
    const float* ln1_g  = (const float*)d_in[1];
    const float* ln1_b  = (const float*)d_in[2];
    const float* qkv_w  = (const float*)d_in[3];
    const float* proj_w = (const float*)d_in[4];
    const float* proj_b = (const float*)d_in[5];
    const float* ln2_g  = (const float*)d_in[6];
    const float* ln2_b  = (const float*)d_in[7];
    const float* fc1_w  = (const float*)d_in[8];
    const float* fc1_b  = (const float*)d_in[9];
    const float* dw_k   = (const float*)d_in[10];
    const float* dw_b   = (const float*)d_in[11];
    const float* fc2_w  = (const float*)d_in[12];
    const float* fc2_b  = (const float*)d_in[13];
    float* out = (float*)d_out;

    char* ws = (char*)d_ws;
    size_t off = 0;
    auto alloc = [&](size_t bytes) {
        void* p = ws + off;
        off += (bytes + 255) & ~(size_t)255;
        return p;
    };
    _Float16* qkvw_t  = (_Float16*)alloc((size_t)DIMC * 3 * DIMC * 2);  // (1152,384)
    _Float16* projw_t = (_Float16*)alloc((size_t)DIMC * DIMC * 2);      // (384,384)
    _Float16* fc1w_t  = (_Float16*)alloc((size_t)DIMC * HID * 2);       // (1536,384)
    _Float16* fc2w_t  = (_Float16*)alloc((size_t)HID * DIMC * 2);       // (384,1536)
    _Float16* xwh     = (_Float16*)alloc((size_t)MROWS * DIMC * 2);     // LN1/LN2 out
    _Float16* qkvh    = (_Float16*)alloc((size_t)MROWS * 3 * DIMC * 2);
    _Float16* attnout = (_Float16*)alloc((size_t)MROWS * DIMC * 2);
    float*    x1      = (float*)   alloc((size_t)MROWS * DIMC * 4);
    _Float16* fc1out  = (_Float16*)alloc((size_t)MROWS * HID * 2);
    _Float16* geluh   = (_Float16*)alloc((size_t)MROWS * HID * 2);

    // 1. weights f32 -> f16, pre-transposed to (N,K) for async B staging
    cvt_transpose<<<(DIMC*3*DIMC + 255)/256, 256, 0, stream>>>(qkv_w,  qkvw_t,  DIMC, 3*DIMC);
    cvt_transpose<<<(DIMC*DIMC   + 255)/256, 256, 0, stream>>>(proj_w, projw_t, DIMC, DIMC);
    cvt_transpose<<<(DIMC*HID    + 255)/256, 256, 0, stream>>>(fc1_w,  fc1w_t,  DIMC, HID);
    cvt_transpose<<<(HID*DIMC    + 255)/256, 256, 0, stream>>>(fc2_w,  fc2w_t,  HID, DIMC);

    // 2. LN1 + roll(-3,-3) + window partition -> f16 (window-ordered rows)
    ln_kernel<1><<<MROWS, 384, 0, stream>>>(x, ln1_g, ln1_b, xwh);

    // 3. QKV GEMM: (50176 x 384) @ (384 x 1152) -> f16
    {
        dim3 g(3*DIMC/BN, MROWS/BM);
        gemm_f16<0><<<g, 256, 0, stream>>>(xwh, qkvw_t, qkvh, nullptr, nullptr,
                                           MROWS, 3*DIMC, DIMC);
    }

    // 4. Windowed attention (per window, per head)
    attn_kernel<<<BB*NWIN*HEADS, 128, 0, stream>>>(qkvh, attnout);

    // 5. proj GEMM + bias + window-reverse + residual(x) -> x1 (f32, spatial order)
    {
        dim3 g(DIMC/BN, MROWS/BM);
        gemm_f16<1><<<g, 256, 0, stream>>>(attnout, projw_t, x1, proj_b, x,
                                           MROWS, DIMC, DIMC);
    }

    // 6. LN2 -> f16 (reuse xwh)
    ln_kernel<0><<<MROWS, 384, 0, stream>>>(x1, ln2_g, ln2_b, xwh);

    // 7. fc1 GEMM + bias -> f16
    {
        dim3 g(HID/BN, MROWS/BM);
        gemm_f16<2><<<g, 256, 0, stream>>>(xwh, fc1w_t, fc1out, fc1_b, nullptr,
                                           MROWS, HID, DIMC);
    }

    // 8. depthwise 3x3 + bias + exact GELU -> f16
    {
        size_t total = (size_t)BB * NSP * HID;
        dwconv_gelu<<<(unsigned)((total + 255) / 256), 256, 0, stream>>>(
            fc1out, dw_k, dw_b, geluh);
    }

    // 9. fc2 GEMM + bias + residual(x1) -> d_out (f32)
    {
        dim3 g(DIMC/BN, MROWS/BM);
        gemm_f16<3><<<g, 256, 0, stream>>>(geluh, fc2w_t, out, fc2_b, x1,
                                           MROWS, DIMC, HID);
    }
}
